// BahdanauAttention_88381837017850
// MI455X (gfx1250) — matrix operand order repaired and checked by reading, not compiled, not run on hardware
//
#include <hip/hip_runtime.h>
#include <hip/hip_bf16.h>
#include <math.h>

// Problem constants
#define B_ 64
#define S_ 4096
#define H_ 256
#define SPLITS 8
#define CHUNK (S_ / SPLITS)   // 512 rows of keys per workgroup
#define NT (CHUNK / 16)       // 32 key tiles per workgroup
#define RSF 260               // LDS key-tile row stride in f32 elems (1040 B; 16B-aligned, banks spread)

typedef __attribute__((ext_vector_type(16))) __bf16 v16bf;
typedef __attribute__((ext_vector_type(8)))  __bf16 bf16x8;
typedef __attribute__((ext_vector_type(8)))  float  v8f;

// Workspace layout (in floats)
#define WS_QP     0
#define WS_SCORES (B_ * H_)                      // 16384
#define WS_PM     (WS_SCORES + B_ * S_)          // 278528
#define WS_PL     (WS_PM + B_ * SPLITS)          // +512
#define WS_PCTX   (WS_PL + B_ * SPLITS)          // +512 ; pctx = 64*8*256 floats

#if __has_builtin(__builtin_amdgcn_global_load_async_to_lds_b128)
#define HAVE_ASYNC_LDS 1
#else
#define HAVE_ASYNC_LDS 0
#endif

#if __has_builtin(__builtin_amdgcn_tanhf)
#define TANH_F32(x) __builtin_amdgcn_tanhf(x)
#else
#define TANH_F32(x) tanhf(x)
#endif

#if HAVE_ASYNC_LDS
// Builtin signature (from diagnostics): param0 = AS1 'int __vector(4)' ptr,
// i.e. (global int4* src, lds int4* dst, imm offset, imm cpol).
typedef int v4i_raw __attribute__((vector_size(16)));
typedef __attribute__((address_space(1))) v4i_raw gbl_v4i;
typedef __attribute__((address_space(3))) v4i_raw lds_v4i;

__device__ __forceinline__ gbl_v4i* as_gbl(const void* p) {
  return (gbl_v4i*)(unsigned long long)p;
}
__device__ __forceinline__ lds_v4i* as_lds(void* p) {
  // generic LDS pointer keeps the LDS byte offset in its low 32 bits
  return (lds_v4i*)(unsigned)(unsigned long long)p;
}

// 64 B per lane (one 16-float row segment) as 4 async b128 copies, VGPR-free.
__device__ __forceinline__ void stage_async(const float* gsrc, float* ldst) {
  __builtin_amdgcn_global_load_async_to_lds_b128(as_gbl(gsrc + 0),  as_lds(ldst + 0),  0, 0);
  __builtin_amdgcn_global_load_async_to_lds_b128(as_gbl(gsrc + 4),  as_lds(ldst + 4),  0, 0);
  __builtin_amdgcn_global_load_async_to_lds_b128(as_gbl(gsrc + 8),  as_lds(ldst + 8),  0, 0);
  __builtin_amdgcn_global_load_async_to_lds_b128(as_gbl(gsrc + 12), as_lds(ldst + 12), 0, 0);
}
#endif

// ---------------------------------------------------------------------------
// Kernel 1: q-projection  qp[b,h] = sum_k query[b,k] * Wq[k,h] + bq[h]
// ---------------------------------------------------------------------------
__global__ void qproj_kernel(const float* __restrict__ query,
                             const float* __restrict__ Wq,
                             const float* __restrict__ bq,
                             float* __restrict__ qp) {
  const int b = blockIdx.x;
  const int h = threadIdx.x;
  float acc = bq[h];
  const float* qrow = query + b * H_;
#pragma unroll 4
  for (int k = 0; k < H_; ++k) acc += qrow[k] * Wq[k * H_ + h];
  qp[b * H_ + h] = acc;
}

// ---------------------------------------------------------------------------
// Kernel 2: fused k-projection (WMMA bf16) + tanh-score + online softmax +
//           weighted-context accumulation.  One pass over keys, with
//           double-buffered async global->LDS staging.
// grid = (SPLITS, B_), block = 256 threads (8 wave32)
// ---------------------------------------------------------------------------
__global__ __launch_bounds__(256, 1)
void attn_main_kernel(const float* __restrict__ keys,
                      const unsigned char* __restrict__ mask,
                      const float* __restrict__ Wk,
                      const float* __restrict__ bk,
                      const float* __restrict__ wsv,
                      const float* __restrict__ bsp,
                      const float* __restrict__ qp,
                      float* __restrict__ scores,
                      float* __restrict__ pm,
                      float* __restrict__ pl,
                      float* __restrict__ pctx) {
  const int split = blockIdx.x;
  const int b     = blockIdx.y;
  const int tid   = threadIdx.x;
  const int lane  = tid & 31;
  const int wave  = tid >> 5;        // 8 waves
  const int half  = lane >> 4;       // 0 or 1
  const int l16   = lane & 15;

  __shared__ __align__(16) float keyTileF[2][16 * RSF];  // double-buffered 16x256 f32
  __shared__ float scoreAcc[16];
  __shared__ float tileScores[16];
  __shared__ float wr_s[16];
  __shared__ float scale_s;

  // --- Preload Wk fragments (B matrix, 32x16 bf16 per fragment) into VGPRs.
  // ISA 16-bit B layout: lane L -> N = L%16;
  // b[0..7] = K ktile*32+8*(L/16)+0..7, b[8..15] = K ktile*32+16+8*(L/16)+0..7.
  v16bf bfrag[2][8];
#pragma unroll
  for (int t = 0; t < 2; ++t) {
    const int n = 32 * wave + 16 * t + l16;
#pragma unroll
    for (int kt = 0; kt < 8; ++kt) {
      const int kb = kt * 32 + 8 * half;
      v16bf f;
#pragma unroll
      for (int i = 0; i < 16; ++i)
        f[i] = (__bf16)Wk[(size_t)(kb + i + (i >= 8 ? 8 : 0)) * H_ + n];
      bfrag[t][kt] = f;
    }
  }

  // Per-lane column constants for the two N-tiles this wave owns.
  float qv[2], wsn[2];
#pragma unroll
  for (int t = 0; t < 2; ++t) {
    const int n = 32 * wave + 16 * t + l16;
    qv[t]  = qp[b * H_ + n] + bk[n];
    wsn[t] = wsv[n];
  }
  const float bsv = bsp[0];

  float run_m = -INFINITY;   // meaningful on thread 0 only
  float run_l = 0.0f;        // meaningful on thread 0 only
  float ctxReg = 0.0f;       // this thread's column (h = tid) of the context

  const int srow = tid >> 4;          // staging: row 0..15
  const int scol = (tid & 15) * 16;   // staging: 16-col segment
  const size_t keyBase = (size_t)b * S_ + split * CHUNK;

#if HAVE_ASYNC_LDS
  // Prologue: start async copy of tile 0
  stage_async(keys + (keyBase + srow) * H_ + scol,
              &keyTileF[0][srow * RSF + scol]);
#endif

  for (int st = 0; st < NT; ++st) {
    const int s0 = split * CHUNK + st * 16;
    float* tile = keyTileF[st & 1];

#if HAVE_ASYNC_LDS
    if (st + 1 < NT) {
      // kick next tile into the other buffer, then wait for current tile
      stage_async(keys + (keyBase + (st + 1) * 16 + srow) * H_ + scol,
                  &keyTileF[(st + 1) & 1][srow * RSF + scol]);
      asm volatile("s_wait_asynccnt 0x4" ::: "memory");
    } else {
      asm volatile("s_wait_asynccnt 0x0" ::: "memory");
    }
#else
    {
      const float4* g = (const float4*)(keys + (keyBase + st * 16 + srow) * H_ + scol);
      float4* d = (float4*)&tile[srow * RSF + scol];
      d[0] = g[0]; d[1] = g[1]; d[2] = g[2]; d[3] = g[3];
    }
#endif
    if (tid < 16) scoreAcc[tid] = 0.0f;
    __syncthreads();

    // ---- k-projection: kp = keysTile @ Wk via 8 WMMA K-steps, 2 N-tiles/wave
    v8f acc0 = {}, acc1 = {};
#pragma unroll
    for (int kt = 0; kt < 8; ++kt) {
      // ISA 16-bit A layout: lane L -> M = L%16; a[0..7]  = K kt*32+8*half+0..7
      //                                          a[8..15] = K kt*32+16+8*half+0..7
      const float* abase = &tile[l16 * RSF + kt * 32 + 8 * half];
      float4 a0 = *(const float4*)(abase + 0);
      float4 a1 = *(const float4*)(abase + 4);
      float4 c0 = *(const float4*)(abase + 16);
      float4 c1 = *(const float4*)(abase + 20);
      v16bf a = {(__bf16)a0.x, (__bf16)a0.y, (__bf16)a0.z, (__bf16)a0.w,
                 (__bf16)a1.x, (__bf16)a1.y, (__bf16)a1.z, (__bf16)a1.w,
                 (__bf16)c0.x, (__bf16)c0.y, (__bf16)c0.z, (__bf16)c0.w,
                 (__bf16)c1.x, (__bf16)c1.y, (__bf16)c1.z, (__bf16)c1.w};
      acc0 = __builtin_amdgcn_wmma_f32_16x16x32_bf16(false, a, false,
                                                     bfrag[0][kt], (short)0,
                                                     acc0, false, false);
      acc1 = __builtin_amdgcn_wmma_f32_16x16x32_bf16(false, a, false,
                                                     bfrag[1][kt], (short)0,
                                                     acc1, false, false);
    }

    // ---- score partials: p = tanh(qp + kp + bk) * ws, reduce over N
    // C/D layout: VGPR v -> row (v + 8*half), col = l16 (+ N-tile offset)
    float rowp[8];
#pragma unroll
    for (int v = 0; v < 8; ++v)
      rowp[v] = TANH_F32(acc0[v] + qv[0]) * wsn[0] +
                TANH_F32(acc1[v] + qv[1]) * wsn[1];
#pragma unroll
    for (int v = 0; v < 8; ++v) {
      float x = rowp[v];
      x += __shfl_xor(x, 1, 16);
      x += __shfl_xor(x, 2, 16);
      x += __shfl_xor(x, 4, 16);
      x += __shfl_xor(x, 8, 16);
      rowp[v] = x;
    }
    if (l16 == 0) {
#pragma unroll
      for (int v = 0; v < 8; ++v)
        atomicAdd(&scoreAcc[v + 8 * half], rowp[v]);  // ds_add_f32
    }
    __syncthreads();

    // ---- online softmax bookkeeping (thread 0)
    if (tid == 0) {
      float tm = -INFINITY;
#pragma unroll
      for (int r = 0; r < 16; ++r) {
        float sv = scoreAcc[r] + bsv;
        if (!mask[(size_t)b * S_ + s0 + r]) sv = -INFINITY;
        tileScores[r] = sv;
        tm = fmaxf(tm, sv);
      }
      const float nm = fmaxf(run_m, tm);
      const float sc = __expf(run_m - nm);
      float ls = 0.0f;
#pragma unroll
      for (int r = 0; r < 16; ++r) {
        const float w = __expf(tileScores[r] - nm);
        wr_s[r] = w;
        ls += w;
      }
      run_l   = run_l * sc + ls;
      run_m   = nm;
      scale_s = sc;
    }
    __syncthreads();

    // ---- emit raw scores; accumulate context column h = tid
    if (tid < 16) scores[(size_t)b * S_ + s0 + tid] = tileScores[tid];
    {
      float a = ctxReg * scale_s;
#pragma unroll
      for (int r = 0; r < 16; ++r)
        a += wr_s[r] * tile[r * RSF + tid];
      ctxReg = a;
    }
    __syncthreads();  // tile buffers / scoreAcc reused next iteration
  }

  const int pidx = b * SPLITS + split;
  if (tid == 0) {
    pm[pidx] = run_m;
    pl[pidx] = run_l;
  }
  pctx[(size_t)pidx * H_ + tid] = ctxReg;
}

// ---------------------------------------------------------------------------
// Kernel 3: combine split partials; write context [B,1,H] then weights [B,S]
// ---------------------------------------------------------------------------
__global__ void finalize_kernel(const float* __restrict__ pm,
                                const float* __restrict__ pl,
                                const float* __restrict__ pctx,
                                const float* __restrict__ scores,
                                float* __restrict__ out) {
  const int b = blockIdx.x;
  const int tid = threadIdx.x;
  __shared__ float coeff[SPLITS];
  __shared__ float Msh, Lsh;
  if (tid == 0) {
    float M = -INFINITY;
    for (int i = 0; i < SPLITS; ++i) M = fmaxf(M, pm[b * SPLITS + i]);
    float L = 0.0f;
    for (int i = 0; i < SPLITS; ++i) {
      const float c = __expf(pm[b * SPLITS + i] - M);
      coeff[i] = c;
      L += c * pl[b * SPLITS + i];
    }
    Msh = M;
    Lsh = L;
  }
  __syncthreads();
  const float invL = 1.0f / Lsh;
  const float M = Msh;

  float ctx = 0.0f;
#pragma unroll
  for (int i = 0; i < SPLITS; ++i)
    ctx += coeff[i] * pctx[(size_t)(b * SPLITS + i) * H_ + tid];
  out[b * H_ + tid] = ctx * invL;

  float* wout = out + B_ * H_;
  for (int s = tid; s < S_; s += 256)
    wout[(size_t)b * S_ + s] = __expf(scores[(size_t)b * S_ + s] - M) * invL;
}

// ---------------------------------------------------------------------------
extern "C" void kernel_launch(void* const* d_in, const int* in_sizes, int n_in,
                              void* d_out, int out_size, void* d_ws,
                              size_t ws_size, hipStream_t stream) {
  const float* query        = (const float*)d_in[0];
  const float* keys         = (const float*)d_in[1];
  const unsigned char* mask = (const unsigned char*)d_in[2];
  const float* Wq           = (const float*)d_in[3];
  const float* bq           = (const float*)d_in[4];
  const float* Wk           = (const float*)d_in[5];
  const float* bk           = (const float*)d_in[6];
  const float* wsv          = (const float*)d_in[7];
  const float* bsp          = (const float*)d_in[8];

  float* ws     = (float*)d_ws;
  float* qp     = ws + WS_QP;
  float* scores = ws + WS_SCORES;
  float* pm     = ws + WS_PM;
  float* pl     = ws + WS_PL;
  float* pctx   = ws + WS_PCTX;
  float* out    = (float*)d_out;

  hipLaunchKernelGGL(qproj_kernel, dim3(B_), dim3(H_), 0, stream,
                     query, Wq, bq, qp);
  hipLaunchKernelGGL(attn_main_kernel, dim3(SPLITS, B_), dim3(256), 0, stream,
                     keys, mask, Wk, bk, wsv, bsp, qp, scores, pm, pl, pctx);
  hipLaunchKernelGGL(finalize_kernel, dim3(B_), dim3(256), 0, stream,
                     pm, pl, pctx, scores, out);
}
